// SpikingNeuralNet_61349312856650
// MI455X (gfx1250) — matrix operand order repaired and checked by reading, not compiled, not run on hardware
//
#include <hip/hip_runtime.h>
#include <hip/hip_bf16.h>
#include <cstdint>
#include <math.h>

// Problem constants (match reference)
#define NN      256     // neurons
#define NS      256     // samples
#define KSP     64      // MAX_SPIKES
#define NSTEPS  100
#define DT0F    0.01f
#define T1F     1.0f
#define VRESET  1.0f
#define ALPHAF  0.03f
#define TWO_PI  6.28318530717958647692f

typedef __attribute__((ext_vector_type(16))) __bf16 v16bf;
typedef __attribute__((ext_vector_type(8)))  float  v8f;

// ---------------- Threefry-2x32 (JAX-compatible, 20 rounds) ----------------
__device__ __forceinline__ void tf2x32(uint32_t k0, uint32_t k1,
                                       uint32_t x0, uint32_t x1,
                                       uint32_t& o0, uint32_t& o1) {
  const uint32_t ks2 = 0x1BD11BDAu ^ k0 ^ k1;
  uint32_t y0 = x0 + k0, y1 = x1 + k1;
#define RND(r) { y0 += y1; y1 = (y1 << (r)) | (y1 >> (32 - (r))); y1 ^= y0; }
  RND(13) RND(15) RND(26) RND(6)
  y0 += k1;  y1 += ks2 + 1u;
  RND(17) RND(29) RND(16) RND(24)
  y0 += ks2; y1 += k0 + 2u;
  RND(13) RND(15) RND(26) RND(6)
  y0 += k0;  y1 += k1 + 3u;
  RND(17) RND(29) RND(16) RND(24)
  y0 += k1;  y1 += ks2 + 4u;
  RND(13) RND(15) RND(26) RND(6)
  y0 += ks2; y1 += k0 + 5u;
#undef RND
  o0 = y0; o1 = y1;
}

// word `idx` of jax random_bits(key,(n,)) with n even: counts split in halves,
// out = concat(out0, out1)
__device__ __forceinline__ uint32_t rbits_word(uint32_t k0, uint32_t k1,
                                               uint32_t idx, uint32_t n) {
  uint32_t half = n >> 1, o0, o1;
  if (idx < half) { tf2x32(k0, k1, idx, idx + half, o0, o1); return o0; }
  tf2x32(k0, k1, idx - half, idx, o0, o1); return o1;
}

__device__ __forceinline__ float bits_to_unit(uint32_t b) {
  return __uint_as_float((b >> 9) | 0x3F800000u) - 1.0f;   // [0,1)
}

__device__ __forceinline__ float uni_eps(uint32_t b) {     // uniform(1e-10, 1)
  return fmaxf(1e-10f, bits_to_unit(b) * (1.0f - 1e-10f) + 1e-10f);
}

// spike_keys = split(fold_in(key(42), 2), 64); key k = words (2k, 2k+1) of
// random_bits over 128 counters
__device__ __forceinline__ void spike_key(int k, uint32_t& sk0, uint32_t& sk1) {
  uint32_t d0, d1;
  tf2x32(0u, 42u, 0u, 2u, d0, d1);             // fold_in(base, 2)
  sk0 = rbits_word(d0, d1, (uint32_t)(2 * k),     128u);
  sk1 = rbits_word(d0, d1, (uint32_t)(2 * k + 1), 128u);
}

// ---------------- init: y0 = (v0, i0, s0), t = 0 ----------------
__global__ void snn_init(const float* __restrict__ v0, const float* __restrict__ i0,
                         float* __restrict__ ws_y, float* __restrict__ ws_t,
                         float* __restrict__ ws_eh) {
  const int s = blockIdx.x, n = threadIdx.x;
  const int idx = s * NN + n;
  uint32_t a0, a1;
  tf2x32(0u, 42u, 0u, 1u, a0, a1);             // fold_in(base, 1)
  uint32_t b = rbits_word(a0, a1, (uint32_t)idx, (uint32_t)(NS * NN));
  float s0v = logf(uni_eps(b)) - ALPHAF;
  ws_y[idx * 3 + 0] = v0[idx];
  ws_y[idx * 3 + 1] = i0[idx];
  ws_y[idx * 3 + 2] = s0v;
  if (n == 0) { ws_t[s] = 0.0f; ws_eh[s] = 0.0f; }
}

// ---------------- w -> bf16, transposed to [n][k] for B-fragment loads ------
__global__ void snn_wprep(const float* __restrict__ w, __bf16* __restrict__ wt) {
  const int kk = blockIdx.x, n = threadIdx.x;
  wt[(size_t)n * NN + kk] = (__bf16)w[(size_t)kk * NN + n];
}

// ---------------- segment: per-sample Euler scan until crossing / T1 --------
__global__ void snn_segment(const float* __restrict__ mu, const float* __restrict__ amp,
                            float* __restrict__ ws_y, float* __restrict__ ws_t,
                            float* __restrict__ ws_eh,
                            float* __restrict__ out_ys, float* __restrict__ out_tev,
                            float* __restrict__ out_et, int k) {
  __shared__ float smin[8];
  const int s = blockIdx.x, n = threadIdx.x;
  const int lane = n & 31, wid = n >> 5;
  const float mu1 = mu[0], mu2 = mu[1];
  const float ia  = amp[n];
  const int   idx = s * NN + n;
  const float INF = __builtin_inff();

  float v  = ws_y[idx * 3 + 0];
  float ii = ws_y[idx * 3 + 1];
  float sv = ws_y[idx * 3 + 2];
  float t  = ws_t[s];                          // per-sample scalar (uniform)

  float tev = t, yev_v = v, yev_i = ii, yev_s = sv;
  bool  mask = false, hit_ever = false;
  bool  done = (t >= T1F - 1e-9f);             // done0

  for (int step = 0; step < NSTEPS && !done; ++step) {
    float ic = ia * (1.0f + sinf(TWO_PI * t));
    float dv = mu1 * (ii + ic - v);
    float di = -mu2 * ii;
    float ds = fmaxf(v, 0.0f) + log1pf(expf(-fabsf(v)));   // stable softplus
    float v_new = v + DT0F * dv, i_new = ii + DT0F * di, s_new = sv + DT0F * ds;
    float t_new = t + DT0F;

    bool  crossed = (s_new >= 0.0f);
    float denom   = crossed ? (s_new - sv) : 1.0f;
    float frac    = crossed ? (-sv / denom) : INF;

    // block-wide min(frac): wave32 shuffle tree + tiny LDS combine
    float m = frac;
    m = fminf(m, __shfl_xor(m, 16, 32));
    m = fminf(m, __shfl_xor(m,  8, 32));
    m = fminf(m, __shfl_xor(m,  4, 32));
    m = fminf(m, __shfl_xor(m,  2, 32));
    m = fminf(m, __shfl_xor(m,  1, 32));
    if (lane == 0) smin[wid] = m;
    __syncthreads();
    float minfrac = smin[0];
#pragma unroll
    for (int q = 1; q < 8; ++q) minfrac = fminf(minfrac, smin[q]);
    __syncthreads();

    bool  anyc = (minfrac < INF);
    float fmin = fminf(fmaxf(minfrac, 0.0f), 1.0f);
    bool  hit     = anyc;                                    // !done inside loop
    bool  reached = (!hit) && (t_new >= T1F - 1e-9f);

    if (hit) {
      tev   = t + fmin * DT0F;
      yev_v = v  + fmin * DT0F * dv;
      yev_i = ii + fmin * DT0F * di;
      yev_s = sv + fmin * DT0F * ds;
      mask  = crossed && (frac <= fmin + 1e-9f);
      hit_ever = true;
      done = true;                                           // uniform exit
    } else if (reached) {
      tev = T1F; yev_v = v_new; yev_i = i_new; yev_s = s_new;
      done = true;
    } else {
      t = t_new; v = v_new; ii = i_new; sv = s_new;
    }
  }

  const long eb = ((long)s * KSP + k) * NN + n;
  out_ys[eb * 3 + 0] = yev_v;
  out_ys[eb * 3 + 1] = yev_i;
  out_ys[eb * 3 + 2] = yev_s;
  out_et[eb] = mask ? 1.0f : 0.0f;
  if (n == 0) {
    out_tev[(long)s * KSP + k] = hit_ever ? tev : INF;
    ws_t[s]  = tev;                                          // carry t = t_ev
    ws_eh[s] = hit_ever ? 1.0f : 0.0f;
  }
}

// ---------------- transition: i += mask @ w via bf16 WMMA, reset v & s ------
// One wave per 16(sample) x 16(neuron) C tile; K loop of 8 x v_wmma 16x16x32.
__global__ void __launch_bounds__(32)
snn_transition(const __bf16* __restrict__ wt, float* __restrict__ ws_y,
               const float* __restrict__ ws_eh,
               const float* __restrict__ out_ys, const float* __restrict__ out_et,
               int k) {
  const int lane = threadIdx.x;                // 0..31
  const int half = lane >> 4, l15 = lane & 15;
  const int n0 = blockIdx.x * 16, s0 = blockIdx.y * 16;

  v8f acc = {};                                // f32 accumulator tile
  for (int k0 = 0; k0 < NN; k0 += 32) {
    v16bf a, b;
    // A fragment (16x32 bf16, row M = s0+l15): lanes 0-15 -> K {0..7,16..23},
    // lanes 16-31 -> K {8..15,24..31}
    const float* mrow = out_et + ((long)(s0 + l15) * KSP + k) * NN;
    const int ka = k0 + (half ? 8 : 0);
#pragma unroll
    for (int e = 0; e < 8; ++e) {
      a[e]     = (__bf16)mrow[ka + e];
      a[8 + e] = (__bf16)mrow[ka + 16 + e];
    }
    // B fragment (32x16 bf16, col N = n0+l15): lanes 0-15 -> K {0..7,16..23},
    // lanes 16-31 -> K {8..15,24..31}; from transposed wt[n][k]
    const __bf16* wrow = wt + (long)(n0 + l15) * NN + k0 + (half ? 8 : 0);
#pragma unroll
    for (int e = 0; e < 16; ++e) b[e] = wrow[e < 8 ? e : e + 8];

    acc = __builtin_amdgcn_wmma_f32_16x16x32_bf16(
        /*neg_a=*/false, a, /*neg_b=*/false, b,
        /*c_mod=*/(short)0, acc, /*reuse_a=*/false, /*reuse_b=*/false);
  }

  uint32_t sk0, sk1;
  spike_key(k, sk0, sk1);

  // C/D layout: VGPR j -> row M = j (lanes 0-15) or j+8 (lanes 16-31), col = lane%16
#pragma unroll
  for (int j = 0; j < 8; ++j) {
    const int srow = s0 + j + half * 8;
    const int ncol = n0 + l15;
    const long eb  = ((long)srow * KSP + k) * NN + ncol;
    const float mk = out_et[eb];
    const float vv = out_ys[eb * 3 + 0];
    const float iv = out_ys[eb * 3 + 1];
    const float ss = out_ys[eb * 3 + 2];
    float vn = vv, in_ = iv, sn = ss;
    if (ws_eh[srow] > 0.5f) {                  // event_happened for this sample
      vn  = vv - VRESET * mk;
      in_ = iv + acc[j];                       // + (mask @ w)[srow][ncol]
      uint32_t b = rbits_word(sk0, sk1, (uint32_t)(srow * NN + ncol),
                              (uint32_t)(NS * NN));
      float s_res = (mk > 0.5f) ? (logf(uni_eps(b)) - ALPHAF) : ss;
      sn = fminf(s_res, -0.001f);
    }
    const long yi = (long)(srow * NN + ncol) * 3;
    ws_y[yi + 0] = vn; ws_y[yi + 1] = in_; ws_y[yi + 2] = sn;
  }
}

// ---------------- host orchestration ----------------
extern "C" void kernel_launch(void* const* d_in, const int* in_sizes, int n_in,
                              void* d_out, int out_size, void* d_ws, size_t ws_size,
                              hipStream_t stream) {
  (void)in_sizes; (void)n_in; (void)out_size; (void)ws_size;
  const float* w   = (const float*)d_in[0];
  const float* mu  = (const float*)d_in[1];
  const float* v0  = (const float*)d_in[2];
  const float* i0  = (const float*)d_in[3];
  const float* amp = (const float*)d_in[4];

  float* out     = (float*)d_out;
  float* out_ys  = out;                                   // (S,K,N,3)
  float* out_tev = out + (long)NS * KSP * NN * 3;         // (S,K)
  float* out_et  = out_tev + (long)NS * KSP;              // (S,K,N)

  char*   ws    = (char*)d_ws;
  float*  ws_y  = (float*)ws;  ws += (size_t)NS * NN * 3 * sizeof(float);
  float*  ws_t  = (float*)ws;  ws += (size_t)NS * sizeof(float);
  float*  ws_eh = (float*)ws;  ws += (size_t)NS * sizeof(float);
  __bf16* wt    = (__bf16*)ws;                            // NN*NN bf16

  snn_init <<<NS, NN, 0, stream>>>(v0, i0, ws_y, ws_t, ws_eh);
  snn_wprep<<<NN, NN, 0, stream>>>(w, wt);
  for (int k = 0; k < KSP; ++k) {
    snn_segment<<<NS, NN, 0, stream>>>(mu, amp, ws_y, ws_t, ws_eh,
                                       out_ys, out_tev, out_et, k);
    snn_transition<<<dim3(NN / 16, NS / 16), 32, 0, stream>>>(
        wt, ws_y, ws_eh, out_ys, out_et, k);
  }
}
